// MsaHmmLayer_17386027614773
// MI455X (gfx1250) — matrix operand order repaired and checked by reading, not compiled, not run on hardware
//
#include <hip/hip_runtime.h>
#include <math.h>

// Problem constants (from reference): NUM_MODEL=2, B=64, T=512, S=26, Q=256
#define NM   2
#define BSZ  64
#define TT   512
#define SS   26
#define QQ   256
#define BT   16          // batch tile per workgroup
#define ASTR 264         // padded alpha row stride (halves): conflict-free frag loads
#define XSTR 40          // padded x-tile row stride (halves), K padded 26 -> 32 (+pad)
#define XBUF (BT * XSTR) // one x-tile buffer, in halves

// ds_swizzle group-of-32 XOR patterns (imm = xor<<10 | or<<5 | and, and=0x1f)
#define SWZ_X1 0x041f
#define SWZ_X2 0x081f
#define SWZ_X4 0x101f
#define SWZ_X8 0x201f

typedef __attribute__((ext_vector_type(16))) _Float16 v16h;
typedef __attribute__((ext_vector_type(2)))  __fp16   v2hf;   // cvt_pkrtz result type
typedef __attribute__((ext_vector_type(8)))  float    v8f;

union FragU { v16h h; unsigned u[8]; };
union PkU   { v2hf h2; unsigned u; };

// packed f16 max without NaN-canonicalization (values are non-NaN by construction)
static __device__ __forceinline__ unsigned pkmax(unsigned a, unsigned b) {
    unsigned d;
    asm("v_pk_max_num_f16 %0, %1, %2" : "=v"(d) : "v"(a), "v"(b));
    return d;
}

// ---- LDS layout (bytes), total 171904 (< 320KB WGP LDS), dynamic shared ----
#define OFF_ALIN   0                    // _Float16 [QQ][QQ]      131072
#define OFF_ALPHA  131072               // _Float16 [BT][ASTR]      8448
#define OFF_BMAT   139520               // float    [QQ][SS]       26624
#define OFF_X16    166144               // _Float16 [2][BT][XSTR]   2560  (double buffer)
#define OFF_PI     168704               // float    [QQ]            1024
#define OFF_PMAX   169728               // unsigned [8][16] packed   512 (1024 reserved)
#define OFF_PSUM   170752               // float    [BT][16]        1024
#define OFF_RSC    171776               // float    [BT]              64
#define OFF_LSC    171840               // float    [BT]              64
#define SMEM_BYTES 171904

__global__ __launch_bounds__(512)
void hmm_forward_kernel(const float* __restrict__ x,
                        const float* __restrict__ tl,
                        const float* __restrict__ el,
                        const float* __restrict__ il,
                        float* __restrict__ out) {
    extern __shared__ char smem[];
    _Float16* A_lin   = (_Float16*)(smem + OFF_ALIN);   // linear transition, row-major [q][p]
    _Float16* alpha16 = (_Float16*)(smem + OFF_ALPHA);  // normalized alpha [b][q], stride ASTR
    float*    Bmat    = (float*)(smem + OFF_BMAT);      // emission probs [q][s]
    _Float16* x16     = (_Float16*)(smem + OFF_X16);    // f16 obs tiles [2][b][s]
    float*    pi_s    = (float*)(smem + OFF_PI);
    unsigned* pmaxu   = (unsigned*)(smem + OFF_PMAX);   // packed f16 row-pair maxes
    float*    psum    = (float*)(smem + OFF_PSUM);
    float*    rscale  = (float*)(smem + OFF_RSC);
    float*    lscale  = (float*)(smem + OFF_LSC);

    const int tid  = threadIdx.x;
    const int m    = blockIdx.x >> 2;          // model index
    const int b0   = (blockIdx.x & 3) * BT;    // batch tile start
    const int lane = tid & 31;
    const int wave = tid >> 5;                 // 16 waves -> 16 N-tiles of 16 cols
    const int n0   = wave * 16;
    const int ln   = lane & 15;
    const int hiL  = lane >> 4;

    // per-thread x staging assignment (tid < 416)
    const int xb = tid / SS;                   // batch row
    const int xs = tid % SS;                   // symbol
    const size_t xg = (((size_t)m * BSZ + b0 + xb) * TT) * SS + xs;

    // ---------------- setup: softmaxes into LDS ----------------
    if (tid < QQ) {
        const float* row = tl + ((size_t)m * QQ + tid) * QQ;
        float mx = -1e30f;
        for (int p = 0; p < QQ; ++p) mx = fmaxf(mx, row[p]);
        float sum = 0.f;
        for (int p = 0; p < QQ; ++p) sum += expf(row[p] - mx);
        float inv = 1.0f / sum;
        for (int p = 0; p < QQ; ++p)
            A_lin[(size_t)tid * QQ + p] = (_Float16)(expf(row[p] - mx) * inv);
        const float* erow = el + ((size_t)m * QQ + tid) * SS;
        float emx = -1e30f;
        for (int s = 0; s < SS; ++s) emx = fmaxf(emx, erow[s]);
        float es = 0.f;
        for (int s = 0; s < SS; ++s) es += expf(erow[s] - emx);
        float einv = 1.0f / es;
        for (int s = 0; s < SS; ++s) Bmat[tid * SS + s] = expf(erow[s] - emx) * einv;
    }
    if (tid == 0) {
        const float* prow = il + (size_t)m * QQ;
        float mx = -1e30f;
        for (int q = 0; q < QQ; ++q) mx = fmaxf(mx, prow[q]);
        float s = 0.f;
        for (int q = 0; q < QQ; ++q) s += expf(prow[q] - mx);
        float inv = 1.0f / s;
        for (int q = 0; q < QQ; ++q) pi_s[q] = expf(prow[q] - mx) * inv;
    }
    if (tid < BT) lscale[tid] = 0.f;
    // zero-fill pad columns s = 26..39 in BOTH x buffers (written once)
    if (tid < 2 * BT * (XSTR - SS)) {
        int buf = tid / (BT * (XSTR - SS));
        int r   = tid % (BT * (XSTR - SS));
        int b = r / (XSTR - SS), s = SS + r % (XSTR - SS);
        x16[buf * XBUF + b * XSTR + s] = (_Float16)0.f;
    }
    // stage t = 0 into buffer 0
    if (tid < BT * SS) x16[xb * XSTR + xs] = (_Float16)x[xg];
    __syncthreads();

    // ---------------- hoist WMMA B-fragments into registers ----------------
    // B layout (32x16 f16, KxN): lane N=ln; vgpr i<4: K=2i+{0,1},
    // i>=4: K=16+2(i-4)+{0,1}; hi half-wave lanes offset K by +8.
    FragU bfrag[8];                         // transition matrix chunks (K=256 in 8x32)
#pragma unroll
    for (int kc = 0; kc < 8; ++kc) {
#pragma unroll
        for (int i = 0; i < 8; ++i) {
            int k = kc * 32 + ((i < 4) ? 0 : 16) + 2 * (i & 3) + hiL * 8;
            unsigned lo = *(const unsigned short*)&A_lin[(size_t)k * QQ + (n0 + ln)];
            unsigned hi = *(const unsigned short*)&A_lin[(size_t)(k + 1) * QQ + (n0 + ln)];
            bfrag[kc].u[i] = lo | (hi << 16);
        }
    }
    FragU bemis;                            // emission matrix B^T: K=s (pad 32), N=q tile
#pragma unroll
    for (int i = 0; i < 8; ++i) {
        int k = ((i < 4) ? 0 : 16) + 2 * (i & 3) + hiL * 8;
        float v0 = (k     < SS) ? Bmat[(n0 + ln) * SS + k]     : 0.f;
        float v1 = (k + 1 < SS) ? Bmat[(n0 + ln) * SS + k + 1] : 0.f;
        _Float16 h0 = (_Float16)v0, h1 = (_Float16)v1;
        unsigned lo = *(unsigned short*)&h0;
        unsigned hi = *(unsigned short*)&h1;
        bemis.u[i] = lo | (hi << 16);
    }
    // Pin invariant fragments into registers: opaque to the compiler so they
    // cannot be rematerialized per iteration.
#pragma unroll
    for (int kc = 0; kc < 8; ++kc) asm volatile("" : "+v"(bfrag[kc].h));
    asm volatile("" : "+v"(bemis.h));

    // emission GEMM: E[b, q-tile] = X (16x32) x B^T (32x16); D layout matches
    // the transition GEMM D layout, so the product stays in registers.
    auto emis = [&](const _Float16* xbuf) -> v8f {
        FragU xf;
#pragma unroll
        for (int i = 0; i < 8; ++i) {
            int k0 = ((i < 4) ? 0 : 16) + 2 * (i & 3) + hiL * 8;
            xf.u[i] = *(const unsigned*)&xbuf[ln * XSTR + k0];
        }
        v8f z = {};
        return __builtin_amdgcn_wmma_f32_16x16x32_f16(
            false, xf.h, false, bemis.h, (short)0, z, false, false);
    };

    // register-resident rescale: v[r] is row b=r+8*hiL, col p=n0+ln of the
    // new alpha (all values nonnegative finite, non-NaN). Row pairs packed to
    // f16x2 (v_cvt_pk_rtz) and reduced across the 16-lane half with ds_swizzle
    // XOR butterflies + raw v_pk_max_num_f16 (no canonicalization). The f16-
    // rounded max is exact for the algorithm: the same mx feeds both rscale
    // and lscale, so the normalization bookkeeping is self-consistent.
    // NO trailing barrier (caller provides the separator).
    auto rescale_reg = [&](v8f v) {
        unsigned pk[4];
#pragma unroll
        for (int j = 0; j < 4; ++j) {
            PkU a;
            a.h2 = __builtin_amdgcn_cvt_pkrtz(v[2 * j], v[2 * j + 1]);
            unsigned u = a.u;
            u = pkmax(u, (unsigned)__builtin_amdgcn_ds_swizzle((int)u, SWZ_X1));
            u = pkmax(u, (unsigned)__builtin_amdgcn_ds_swizzle((int)u, SWZ_X2));
            u = pkmax(u, (unsigned)__builtin_amdgcn_ds_swizzle((int)u, SWZ_X4));
            u = pkmax(u, (unsigned)__builtin_amdgcn_ds_swizzle((int)u, SWZ_X8));
            pk[j] = u;
        }
        if (ln == 0) {
#pragma unroll
            for (int j = 0; j < 4; ++j)
                pmaxu[(j + 4 * hiL) * 16 + wave] = pk[j];   // rows (2j[+8hiL]), (2j+1[+8hiL])
        }
        __syncthreads();
        if (tid < BT) {
            int b = tid;
            int jp   = ((b & 7) >> 1) + 4 * (b >> 3);       // row-pair slot
            int half = b & 1;                                // low/high f16 of the pair
            unsigned mu = 0;
            for (int c = 0; c < 16; ++c) {
                unsigned h = (pmaxu[jp * 16 + c] >> (16 * half)) & 0xffffu;
                mu = (h > mu) ? h : mu;                      // uint order == f16 order (nonneg)
            }
            unsigned short hs = (unsigned short)mu;
            _Float16 hf = *(_Float16*)&hs;
            float mx = fmaxf((float)hf, 1e-30f);
            rscale[tid] = 1.0f / mx;
            lscale[tid] += logf(mx);
        }
        __syncthreads();
        // pack row pairs (4x cvt_pk instead of 8x cvt), store halves b16/b16_hi
#pragma unroll
        for (int j = 0; j < 4; ++j) {
            int bA = 2 * j + 8 * hiL, bB = bA + 1;
            PkU p;
            p.h2 = __builtin_amdgcn_cvt_pkrtz(v[2 * j] * rscale[bA],
                                              v[2 * j + 1] * rscale[bB]);
            *(unsigned short*)&alpha16[bA * ASTR + n0 + ln] = (unsigned short)(p.u & 0xffffu);
            *(unsigned short*)&alpha16[bB * ASTR + n0 + ln] = (unsigned short)(p.u >> 16);
        }
    };

    // ---------------- t = 0: alpha0 = pi * E0 (+ prefetch t=1) ----------------
    {
        float xpre = 0.f;
        if (tid < BT * SS) xpre = x[xg + (size_t)SS];     // t = 1
        v8f e0 = emis(x16);
        float piv = pi_s[n0 + ln];
        v8f v;
#pragma unroll
        for (int r = 0; r < 8; ++r) v[r] = e0[r] * piv;
        // store prefetched tile into buffer 1 before the pmax barrier
        if (tid < BT * SS) x16[XBUF + xb * XSTR + xs] = (_Float16)xpre;
        rescale_reg(v);
    }

    // ---------------- forward recursion ----------------
    for (int t = 1; t < TT; ++t) {
        __syncthreads();   // prev step's alpha16 / x16-next writes -> this step's reads

        // prefetch observation tile for t+1 (latency hidden behind WMMA chain)
        float xpre = 0.f;
        if (tid < BT * SS) {
            int tn = (t + 1 < TT) ? (t + 1) : t;
            xpre = x[xg + (size_t)tn * SS];
        }

        const _Float16* xcur = x16 + (t & 1) * XBUF;
        v8f acc_e = emis(xcur);            // emission in registers, D layout

        // GEMM: new[b, p] = sum_q alpha[b, q] * A[q, p]  (16x256 @ 256x256)
        v8f acc = {};
#pragma unroll
        for (int kc = 0; kc < 8; ++kc) {
            // A layout (16x32 f16, MxK): lane M=ln; vgpr i<4: K=2i+{0,1},
            // i>=4: K=16+2(i-4)+{0,1}; hi half-wave lanes offset K by +8.
            FragU af;
#pragma unroll
            for (int i = 0; i < 8; ++i) {
                int k0 = kc * 32 + ((i < 4) ? 0 : 16) + 2 * (i & 3) + hiL * 8;
                af.u[i] = *(const unsigned*)&alpha16[ln * ASTR + k0];
            }
            acc = __builtin_amdgcn_wmma_f32_16x16x32_f16(
                false, af.h, false, bfrag[kc].h, (short)0, acc, false, false);
        }

        // fuse emission multiply entirely in registers (matching D layouts)
        v8f v;
#pragma unroll
        for (int r = 0; r < 8; ++r) v[r] = acc[r] * acc_e[r];

        // store prefetched tile into the other buffer (disjoint from xcur)
        if (tid < BT * SS) x16[((t + 1) & 1) * XBUF + xb * XSTR + xs] = (_Float16)xpre;

        rescale_reg(v);
    }

    // ---------------- finalize: loglik = log(sum alpha_norm) + log_scale ----------------
    __syncthreads();
    if (tid < QQ) {
        int b = tid >> 4, c = tid & 15;
        float s = 0.f;
        const _Float16* r = alpha16 + b * ASTR + c * 16;
        for (int k = 0; k < 16; ++k) s += (float)r[k];
        psum[b * 16 + c] = s;
    }
    __syncthreads();
    if (tid < BT) {
        float s = 0.f;
        for (int c = 0; c < 16; ++c) s += psum[tid * 16 + c];
        out[(size_t)m * BSZ + b0 + tid] = logf(s) + lscale[tid];
    }
}

extern "C" void kernel_launch(void* const* d_in, const int* in_sizes, int n_in,
                              void* d_out, int out_size, void* d_ws, size_t ws_size,
                              hipStream_t stream) {
    (void)in_sizes; (void)n_in; (void)out_size; (void)d_ws; (void)ws_size;
    const float* x  = (const float*)d_in[0];   // (2, 64, 512, 26)
    const float* tl = (const float*)d_in[1];   // (2, 256, 256)
    const float* el = (const float*)d_in[2];   // (2, 256, 26)
    const float* il = (const float*)d_in[3];   // (2, 256)
    float* out = (float*)d_out;                // (2, 64)

    dim3 grid(NM * (BSZ / BT));   // 8 workgroups: (model, batch tile)
    dim3 block(512);              // 16 wave32 -> 16 N-tiles of the 256-wide state dim
    hipLaunchKernelGGL(hmm_forward_kernel, grid, block, SMEM_BYTES, stream,
                       x, tl, el, il, out);
}
